// LinearAttention_89850715833062
// MI455X (gfx1250) — matrix operand order repaired and checked by reading, not compiled, not run on hardware
//
#include <hip/hip_runtime.h>

// ---------------------------------------------------------------------------
// Linear attention on MI455X (gfx1250): bf16 WMMA pipeline, f32 accumulation.
// ---------------------------------------------------------------------------

typedef __attribute__((ext_vector_type(16))) __bf16 v16bf;
typedef __attribute__((ext_vector_type(8)))  float  v8f;

union Frag {
    v16bf v;
    uint4 u[2];
};

#define BATCH  4
#define SEQ    4096
#define CH     768
#define HEADS  12
#define HDIM   64
#define MROWS  (BATCH * SEQ)      // 16384
#define C3     (3 * CH)           // 2304
#define LDT    72                 // padded LDS row stride (ushorts), 144B = 9*16B

static __device__ __forceinline__ unsigned short f2bf(float f) {
    unsigned int u = __float_as_uint(f);
    u += 0x7FFFu + ((u >> 16) & 1u);          // round-to-nearest-even
    return (unsigned short)(u >> 16);
}
static __device__ __forceinline__ float bf2f(unsigned int h) {
    return __uint_as_float((h & 0xFFFFu) << 16);
}
static __device__ __forceinline__ float phi_map(float x) {
    return x > 0.0f ? x + 1.0f : __expf(x);   // elu(x)+1
}
static __device__ __forceinline__ v8f wmma_bf(const Frag& a, const Frag& b, v8f c) {
    return __builtin_amdgcn_wmma_f32_16x16x32_bf16(
        false, a.v, false, b.v, (short)0, c, false, false);
}

// ---------------------------------------------------------------------------
// K0: fp32 -> bf16 conversion
// ---------------------------------------------------------------------------
__global__ void convert_bf16_kernel(const float* __restrict__ src,
                                    unsigned short* __restrict__ dst, int n) {
    int i = blockIdx.x * blockDim.x + threadIdx.x;
    if (i < n) dst[i] = f2bf(src[i]);
}

// ---------------------------------------------------------------------------
// K1: QKV GEMM  C[m,n] = sum_k x_bf[m,k] * Wqkv_bf[n,k]
//     M=16384, N=2304, K=768.  Block tile 128x128x64, 256 threads (8 waves).
//     Epilogue: phi on q,k; q -> [B,H,N,d]; k,v -> transposed [B,H,d,N].
// ---------------------------------------------------------------------------
__global__ __launch_bounds__(256) void qkv_gemm_kernel(
        const unsigned short* __restrict__ Abf,   // [16384,768]
        const unsigned short* __restrict__ Wbf,   // [2304,768]
        unsigned short* __restrict__ qbuf,        // [B,H,N,d]
        unsigned short* __restrict__ kT,          // [B,H,d,N]
        unsigned short* __restrict__ vT) {        // [B,H,d,N]
    __shared__ unsigned short lds_a[128 * LDT];
    __shared__ unsigned short lds_b[128 * LDT];

    const int tid  = threadIdx.x;
    const int lane = tid & 31, w = tid >> 5;
    const int lh = lane >> 4, lr = lane & 15;
    const int wrow = w >> 2, wcol = w & 3;        // 2 x 4 wave grid
    const int m0 = blockIdx.y * 128;
    const int n0 = blockIdx.x * 128;

    v8f zero = {};
    v8f acc[4][2];
    #pragma unroll
    for (int tr = 0; tr < 4; ++tr)
        #pragma unroll
        for (int tc = 0; tc < 2; ++tc) acc[tr][tc] = zero;

    for (int k0 = 0; k0 < CH; k0 += 64) {
        __syncthreads();
        #pragma unroll
        for (int i = 0; i < 4; ++i) {
            int idx = tid + i * 256;              // 1024 16B chunks per tile
            int r = idx >> 3, cc = (idx & 7) * 8;
            *(uint4*)&lds_a[r * LDT + cc] =
                *(const uint4*)&Abf[(size_t)(m0 + r) * CH + k0 + cc];
            *(uint4*)&lds_b[r * LDT + cc] =
                *(const uint4*)&Wbf[(size_t)(n0 + r) * CH + k0 + cc];
        }
        __syncthreads();
        #pragma unroll
        for (int kk = 0; kk < 64; kk += 32) {
            Frag a[4], bb[2];
            #pragma unroll
            for (int tr = 0; tr < 4; ++tr) {
                const unsigned short* p =
                    &lds_a[(wrow * 64 + tr * 16 + lr) * LDT + kk + lh * 8];
                a[tr].u[0] = *(const uint4*)p;
                a[tr].u[1] = *(const uint4*)(p + 16);
            }
            #pragma unroll
            for (int tc = 0; tc < 2; ++tc) {
                const unsigned short* p =
                    &lds_b[(wcol * 32 + tc * 16 + lr) * LDT + kk + lh * 8];
                bb[tc].u[0] = *(const uint4*)p;
                bb[tc].u[1] = *(const uint4*)(p + 16);
            }
            #pragma unroll
            for (int tr = 0; tr < 4; ++tr)
                #pragma unroll
                for (int tc = 0; tc < 2; ++tc)
                    acc[tr][tc] = wmma_bf(a[tr], bb[tc], acc[tr][tc]);
        }
    }

    const int s = n0 / CH;                         // 0=q 1=k 2=v, uniform per block
    #pragma unroll
    for (int tr = 0; tr < 4; ++tr) {
        #pragma unroll
        for (int tc = 0; tc < 2; ++tc) {
            int gnc = (n0 % CH) + wcol * 32 + tc * 16 + lr;   // 0..767
            int h = gnc >> 6, dd = gnc & 63;
            #pragma unroll
            for (int i = 0; i < 8; ++i) {
                int gm = m0 + wrow * 64 + tr * 16 + i + lh * 8;
                int b = gm >> 12, nseq = gm & 4095;
                float val = acc[tr][tc][i];
                size_t bh = (size_t)(b * HEADS + h);
                if (s == 0) {
                    qbuf[(bh * SEQ + nseq) * HDIM + dd] = f2bf(phi_map(val));
                } else if (s == 1) {
                    kT[(bh * HDIM + dd) * SEQ + nseq] = f2bf(phi_map(val));
                } else {
                    vT[(bh * HDIM + dd) * SEQ + nseq] = f2bf(val);
                }
            }
        }
    }
}

// ---------------------------------------------------------------------------
// K2: per (b,h): S[dk,m] = sum_n k[n,dk] v[n,m]  (stored transposed: S_T[m,dk])
//     plus k_sum[dk] = sum_n k[n,dk].  Waves 0-3: WMMA;  waves 4-5: k_sum.
// ---------------------------------------------------------------------------
__global__ __launch_bounds__(256) void kv_ksum_kernel(
        const unsigned short* __restrict__ kT,    // [BH, d, N]
        const unsigned short* __restrict__ vT,    // [BH, d, N]
        unsigned short* __restrict__ S_T,         // [BH, m, dk] bf16
        float* __restrict__ ksum) {               // [BH, d]
    const int bh = blockIdx.x;
    const int tid = threadIdx.x, lane = tid & 31, w = tid >> 5;
    const int lh = lane >> 4, lr = lane & 15;
    const unsigned short* kbase = kT + (size_t)bh * HDIM * SEQ;
    const unsigned short* vbase = vT + (size_t)bh * HDIM * SEQ;

    if (w < 4) {
        v8f zero = {};
        v8f acc[4];
        #pragma unroll
        for (int tc = 0; tc < 4; ++tc) acc[tc] = zero;
        for (int n0 = 0; n0 < SEQ; n0 += 32) {
            Frag a, bb[4];
            const unsigned short* pa =
                kbase + (size_t)(w * 16 + lr) * SEQ + n0 + lh * 8;
            a.u[0] = *(const uint4*)pa;
            a.u[1] = *(const uint4*)(pa + 16);
            #pragma unroll
            for (int tc = 0; tc < 4; ++tc) {
                const unsigned short* pb =
                    vbase + (size_t)(tc * 16 + lr) * SEQ + n0 + lh * 8;
                bb[tc].u[0] = *(const uint4*)pb;
                bb[tc].u[1] = *(const uint4*)(pb + 16);
            }
            #pragma unroll
            for (int tc = 0; tc < 4; ++tc) acc[tc] = wmma_bf(a, bb[tc], acc[tc]);
        }
        #pragma unroll
        for (int tc = 0; tc < 4; ++tc)
            #pragma unroll
            for (int i = 0; i < 8; ++i) {
                int dk = w * 16 + i + lh * 8;
                int m  = tc * 16 + lr;
                S_T[((size_t)bh * HDIM + m) * HDIM + dk] = f2bf(acc[tc][i]);
            }
    } else {
        int idx = (w - 4) * 32 + lane;            // 0..127; rows 0..63 active
        if (idx < HDIM) {
            const unsigned short* p = kbase + (size_t)idx * SEQ;
            float s = 0.0f;
            for (int n = 0; n < SEQ; n += 8) {
                uint4 u = *(const uint4*)(p + n);
                s += bf2f(u.x) + bf2f(u.x >> 16) + bf2f(u.y) + bf2f(u.y >> 16) +
                     bf2f(u.z) + bf2f(u.z >> 16) + bf2f(u.w) + bf2f(u.w >> 16);
            }
            ksum[bh * HDIM + idx] = s;
        }
    }
}

// ---------------------------------------------------------------------------
// K2.5: z[bh,n] = 1 / (q[bh,n,:] . ksum[bh,:] + eps)
// ---------------------------------------------------------------------------
__global__ void z_kernel(const unsigned short* __restrict__ qbuf,
                         const float* __restrict__ ksum,
                         float* __restrict__ z) {
    int t = blockIdx.x * blockDim.x + threadIdx.x;
    if (t >= BATCH * HEADS * SEQ) return;
    int bh = t >> 12;
    const unsigned short* qp = qbuf + (size_t)t * HDIM;
    const float* kp = ksum + bh * HDIM;
    float s = 0.0f;
    #pragma unroll
    for (int j = 0; j < HDIM; j += 8) {
        uint4 u = *(const uint4*)(qp + j);
        s += bf2f(u.x)       * kp[j + 0] + bf2f(u.x >> 16) * kp[j + 1];
        s += bf2f(u.y)       * kp[j + 2] + bf2f(u.y >> 16) * kp[j + 3];
        s += bf2f(u.z)       * kp[j + 4] + bf2f(u.z >> 16) * kp[j + 5];
        s += bf2f(u.w)       * kp[j + 6] + bf2f(u.w >> 16) * kp[j + 7];
    }
    z[t] = 1.0f / (s + 1e-6f);
}

// ---------------------------------------------------------------------------
// K3: y[n,m] = z[n] * sum_dk q[n,dk] S[dk,m];  write bf16 to [B,N,C] layout.
//     grid (48, 32); 8 waves, each one 16-row tile x 64 cols.
// ---------------------------------------------------------------------------
__global__ __launch_bounds__(256) void y_kernel(
        const unsigned short* __restrict__ qbuf,  // [BH, N, d]
        const unsigned short* __restrict__ S_T,   // [BH, m, dk]
        const float* __restrict__ z,              // [BH, N]
        unsigned short* __restrict__ ybf) {       // [B, N, C]
    const int bh = blockIdx.x;
    const int b = bh / HEADS, h = bh % HEADS;
    const int tid = threadIdx.x, lane = tid & 31, w = tid >> 5;
    const int lh = lane >> 4, lr = lane & 15;
    const int nrow0 = blockIdx.y * 128 + w * 16;

    v8f zero = {};
    v8f acc[4];
    #pragma unroll
    for (int tc = 0; tc < 4; ++tc) acc[tc] = zero;

    #pragma unroll
    for (int kk = 0; kk < HDIM; kk += 32) {
        Frag a, bb[4];
        const unsigned short* pa =
            qbuf + ((size_t)bh * SEQ + nrow0 + lr) * HDIM + kk + lh * 8;
        a.u[0] = *(const uint4*)pa;
        a.u[1] = *(const uint4*)(pa + 16);
        #pragma unroll
        for (int tc = 0; tc < 4; ++tc) {
            const unsigned short* pb =
                S_T + ((size_t)bh * HDIM + tc * 16 + lr) * HDIM + kk + lh * 8;
            bb[tc].u[0] = *(const uint4*)pb;
            bb[tc].u[1] = *(const uint4*)(pb + 16);
        }
        #pragma unroll
        for (int tc = 0; tc < 4; ++tc) acc[tc] = wmma_bf(a, bb[tc], acc[tc]);
    }

    #pragma unroll
    for (int i = 0; i < 8; ++i) {
        int n = nrow0 + i + lh * 8;
        float zv = z[(size_t)bh * SEQ + n];
        #pragma unroll
        for (int tc = 0; tc < 4; ++tc) {
            int m = tc * 16 + lr;
            ybf[((size_t)(b * SEQ + n)) * CH + h * HDIM + m] =
                f2bf(acc[tc][i] * zv);
        }
    }
}

// ---------------------------------------------------------------------------
// K4: out[m,n] = sum_k y_bf[m,k] * Wproj_bf[n,k] + bias[n]   (fp32 output)
//     M=16384, N=768, K=768.  Same tiling as K1.
// ---------------------------------------------------------------------------
__global__ __launch_bounds__(256) void proj_gemm_kernel(
        const unsigned short* __restrict__ Ybf,   // [16384,768]
        const unsigned short* __restrict__ Wbf,   // [768,768]
        const float* __restrict__ bias,           // [768]
        float* __restrict__ out) {                // [16384,768]
    __shared__ unsigned short lds_a[128 * LDT];
    __shared__ unsigned short lds_b[128 * LDT];

    const int tid  = threadIdx.x;
    const int lane = tid & 31, w = tid >> 5;
    const int lh = lane >> 4, lr = lane & 15;
    const int wrow = w >> 2, wcol = w & 3;
    const int m0 = blockIdx.y * 128;
    const int n0 = blockIdx.x * 128;

    v8f zero = {};
    v8f acc[4][2];
    #pragma unroll
    for (int tr = 0; tr < 4; ++tr)
        #pragma unroll
        for (int tc = 0; tc < 2; ++tc) acc[tr][tc] = zero;

    for (int k0 = 0; k0 < CH; k0 += 64) {
        __syncthreads();
        #pragma unroll
        for (int i = 0; i < 4; ++i) {
            int idx = tid + i * 256;
            int r = idx >> 3, cc = (idx & 7) * 8;
            *(uint4*)&lds_a[r * LDT + cc] =
                *(const uint4*)&Ybf[(size_t)(m0 + r) * CH + k0 + cc];
            *(uint4*)&lds_b[r * LDT + cc] =
                *(const uint4*)&Wbf[(size_t)(n0 + r) * CH + k0 + cc];
        }
        __syncthreads();
        #pragma unroll
        for (int kk = 0; kk < 64; kk += 32) {
            Frag a[4], bb[2];
            #pragma unroll
            for (int tr = 0; tr < 4; ++tr) {
                const unsigned short* p =
                    &lds_a[(wrow * 64 + tr * 16 + lr) * LDT + kk + lh * 8];
                a[tr].u[0] = *(const uint4*)p;
                a[tr].u[1] = *(const uint4*)(p + 16);
            }
            #pragma unroll
            for (int tc = 0; tc < 2; ++tc) {
                const unsigned short* p =
                    &lds_b[(wcol * 32 + tc * 16 + lr) * LDT + kk + lh * 8];
                bb[tc].u[0] = *(const uint4*)p;
                bb[tc].u[1] = *(const uint4*)(p + 16);
            }
            #pragma unroll
            for (int tr = 0; tr < 4; ++tr)
                #pragma unroll
                for (int tc = 0; tc < 2; ++tc)
                    acc[tr][tc] = wmma_bf(a[tr], bb[tc], acc[tr][tc]);
        }
    }

    #pragma unroll
    for (int tr = 0; tr < 4; ++tr) {
        #pragma unroll
        for (int tc = 0; tc < 2; ++tc) {
            int gn = n0 + wcol * 32 + tc * 16 + lr;
            float bv = bias[gn];
            #pragma unroll
            for (int i = 0; i < 8; ++i) {
                int gm = m0 + wrow * 64 + tr * 16 + i + lh * 8;
                out[(size_t)gm * CH + gn] = acc[tr][tc][i] + bv;
            }
        }
    }
}

// ---------------------------------------------------------------------------
// Launcher
// ---------------------------------------------------------------------------
extern "C" void kernel_launch(void* const* d_in, const int* in_sizes, int n_in,
                              void* d_out, int out_size, void* d_ws, size_t ws_size,
                              hipStream_t stream) {
    const float* x     = (const float*)d_in[0];   // [4,4096,768]
    const float* Wqkv  = (const float*)d_in[1];   // [2304,768]
    const float* Wproj = (const float*)d_in[2];   // [768,768]
    const float* bproj = (const float*)d_in[3];   // [768]
    float* out = (float*)d_out;

    // workspace layout (bytes); x_bf buffer is reused for y_bf after K1.
    char* ws = (char*)d_ws;
    size_t off = 0;
    unsigned short* x_bf    = (unsigned short*)(ws + off); off += (size_t)MROWS * CH * 2;       // 25.2 MB (also y_bf)
    unsigned short* wqkv_bf = (unsigned short*)(ws + off); off += (size_t)C3 * CH * 2;          //  3.5 MB
    unsigned short* wprj_bf = (unsigned short*)(ws + off); off += (size_t)CH * CH * 2;          //  1.2 MB
    unsigned short* qbuf    = (unsigned short*)(ws + off); off += (size_t)MROWS * CH * 2 / 3 * 0 + (size_t)BATCH * HEADS * SEQ * HDIM * 2; // 25.2 MB
    unsigned short* kTbuf   = (unsigned short*)(ws + off); off += (size_t)BATCH * HEADS * SEQ * HDIM * 2;   // 25.2 MB
    unsigned short* vTbuf   = (unsigned short*)(ws + off); off += (size_t)BATCH * HEADS * SEQ * HDIM * 2;   // 25.2 MB
    unsigned short* S_T     = (unsigned short*)(ws + off); off += (size_t)BATCH * HEADS * HDIM * HDIM * 2;  //  0.4 MB
    float*          ksum    = (float*)(ws + off);          off += (size_t)BATCH * HEADS * HDIM * 4;
    float*          zbuf    = (float*)(ws + off);          off += (size_t)BATCH * HEADS * SEQ * 4;
    unsigned short* y_bf    = x_bf;   // reuse (x_bf dead after K1)

    // K0: conversions
    {
        int nx = MROWS * CH;            // 12,582,912
        int nq = C3 * CH;               //  1,769,472
        int np = CH * CH;               //    589,824
        convert_bf16_kernel<<<(nx + 255) / 256, 256, 0, stream>>>(x, x_bf, nx);
        convert_bf16_kernel<<<(nq + 255) / 256, 256, 0, stream>>>(Wqkv, wqkv_bf, nq);
        convert_bf16_kernel<<<(np + 255) / 256, 256, 0, stream>>>(Wproj, wprj_bf, np);
    }

    // K1: QKV projection (M=16384, N=2304): grid (18, 128)
    qkv_gemm_kernel<<<dim3(C3 / 128, MROWS / 128), 256, 0, stream>>>(
        x_bf, wqkv_bf, qbuf, kTbuf, vTbuf);

    // K2: kv outer-product + k_sum: one WG per (b,h)
    kv_ksum_kernel<<<BATCH * HEADS, 256, 0, stream>>>(kTbuf, vTbuf, S_T, ksum);

    // K2.5: normalizer z
    z_kernel<<<(BATCH * HEADS * SEQ + 255) / 256, 256, 0, stream>>>(qbuf, ksum, zbuf);

    // K3: y = phi(q) @ S * z  -> bf16 [B,N,C]
    y_kernel<<<dim3(BATCH * HEADS, SEQ / 128), 256, 0, stream>>>(qbuf, S_T, zbuf, y_bf);

    // K4: output projection + bias -> fp32 out: grid (6, 128)
    proj_gemm_kernel<<<dim3(CH / 128, MROWS / 128), 256, 0, stream>>>(
        y_bf, wprj_bf, bproj, out);
}
